// MultipleHeadSelfAttention_28321014350476
// MI455X (gfx1250) — matrix operand order repaired and checked by reading, not compiled, not run on hardware
//
#include <hip/hip_runtime.h>
#include <math.h>

// Multi-head self-attention, CDNA5 (gfx1250) wave32 + v_wmma_f32_16x16x32_bf16.
// B=128 T=256 C=384 H=6 Dh=64.

#define BATCH 128
#define SEQ   256
#define EMB   384
#define NHEAD 6
#define HDIM  64
#define EMB3  (3*EMB)
#define ROWS  (BATCH*SEQ)          // 32768
#define SOFTMAX_SCALE 0.125f       // 1/sqrt(64)

typedef __attribute__((ext_vector_type(16))) __bf16  v16bf;
typedef __attribute__((ext_vector_type(8)))  float   v8f;
typedef __attribute__((ext_vector_type(8)))  unsigned short u16x8;

union FragU { u16x8 u[2]; v16bf v; };

static __device__ __forceinline__ v16bf load_frag(const unsigned short* p0,
                                                  const unsigned short* p1) {
    FragU f;
    f.u[0] = *(const u16x8*)p0;
    f.u[1] = *(const u16x8*)p1;
    return f.v;
}

static __device__ __forceinline__ unsigned short f32_to_bf16(float f) {
    unsigned int u = __float_as_uint(f);
    unsigned int lsb = (u >> 16) & 1u;
    u += 0x7fffu + lsb;               // round to nearest even
    return (unsigned short)(u >> 16);
}

static __device__ __forceinline__ v8f wmma_bf16(v16bf a, v16bf b, v8f c) {
    return __builtin_amdgcn_wmma_f32_16x16x32_bf16(false, a, false, b,
                                                   (short)0, c, false, false);
}

// ---------------------------------------------------------------- conversions
__global__ void k_f32_to_bf16(const float* __restrict__ in,
                              unsigned short* __restrict__ out, int n) {
    int i = blockIdx.x * blockDim.x + threadIdx.x;
    if (i < n) out[i] = f32_to_bf16(in[i]);
}

// in[rows][cols] fp32 -> out[cols][rows] bf16  (so GEMM B-frags are contiguous)
__global__ void k_transpose_bf16(const float* __restrict__ in,
                                 unsigned short* __restrict__ out,
                                 int rows, int cols) {
    int i = blockIdx.x * blockDim.x + threadIdx.x;
    if (i >= rows * cols) return;
    int r = i / cols, c = i - r * cols;
    out[(size_t)c * rows + r] = f32_to_bf16(in[i]);
}

// ---------------------------------------------------------------- QKV GEMM
// xb[ROWS][EMB] (bf16) @ Wt[EMB3][EMB] (bf16, pre-transposed).
// Each wave computes a 16x64 output tile (4 WMMAs share one A-fragment), then
// scatters into Q,K [B,H,T,Dh] and V transposed [B,H,Dh,T], all bf16.
__global__ void __launch_bounds__(256)
k_qkv_gemm(const unsigned short* __restrict__ xb,
           const unsigned short* __restrict__ wT,
           unsigned short* __restrict__ Qb,
           unsigned short* __restrict__ Kb,
           unsigned short* __restrict__ Vtb) {
    const int NCT = EMB3 / 64;                       // 18 col super-tiles
    int wave = (blockIdx.x * blockDim.x + threadIdx.x) >> 5;
    int lane = threadIdx.x & 31;
    int tr = wave / NCT, tc = wave - tr * NCT;
    if (tr >= ROWS / 16) return;                     // wave-uniform
    int half = lane >> 4, l16 = lane & 15;
    int ka0 = half * 16, ka1 = half * 16 + 8, kb2 = half * 16;

    const unsigned short* arow = xb + (size_t)(tr * 16 + l16) * EMB;
    const unsigned short* bc0  = wT + (size_t)(tc * 64 + l16) * EMB + kb2;

    v8f acc[4] = {{}, {}, {}, {}};
#pragma unroll
    for (int k = 0; k < EMB; k += 32) {
        v16bf a = load_frag(arow + k + ka0, arow + k + ka1);
#pragma unroll
        for (int n = 0; n < 4; ++n) {
            const unsigned short* bp = bc0 + (size_t)(n * 16) * EMB + k;
            acc[n] = wmma_bf16(a, load_frag(bp, bp + 8), acc[n]);
        }
    }

#pragma unroll
    for (int n = 0; n < 4; ++n) {
        int col = tc * 64 + n * 16 + l16;
        int which = col / EMB, rem = col - which * EMB;
        int h = rem / HDIM, dh = rem - h * HDIM;
#pragma unroll
        for (int r = 0; r < 8; ++r) {
            int row = tr * 16 + r + half * 8;        // global token index
            int bi = row >> 8, t = row & 255;
            unsigned short v = f32_to_bf16(acc[n][r]);
            size_t bh = (size_t)bi * NHEAD + h;
            if (which == 0)      Qb [(bh * SEQ + t) * HDIM + dh] = v;
            else if (which == 1) Kb [(bh * SEQ + t) * HDIM + dh] = v;
            else                 Vtb[(bh * HDIM + dh) * SEQ + t] = v;
        }
    }
}

// ---------------------------------------------------------------- flash attn
// One wave handles one (b, h, 16-query tile). Causal, online softmax.
__global__ void __launch_bounds__(256)
k_attention(const unsigned short* __restrict__ Qb,
            const unsigned short* __restrict__ Kb,
            const unsigned short* __restrict__ Vtb,
            unsigned short* __restrict__ attnb) {
    __shared__ unsigned short ldsP[8][16 * 32];      // per-wave P tile (8 KB)

    int wib  = threadIdx.x >> 5;
    int wave = blockIdx.x * 8 + wib;
    int lane = threadIdx.x & 31;
    int half = lane >> 4, l16 = lane & 15;
    int qt = wave & 15, bh = wave >> 4;
    int h = bh % NHEAD, b = bh / NHEAD;
    if (b >= BATCH) return;                          // wave-uniform
    int q0 = qt * 16;

    const unsigned short* Q  = Qb  + (size_t)(b * NHEAD + h) * SEQ * HDIM;
    const unsigned short* K  = Kb  + (size_t)(b * NHEAD + h) * SEQ * HDIM;
    const unsigned short* Vt = Vtb + (size_t)(b * NHEAD + h) * HDIM * SEQ;
    unsigned short* P = ldsP[wib];

    int ka0 = half * 16, ka1 = half * 16 + 8, kb2 = half * 16;

    // Q fragments are loop-invariant for this wave: hoist (16 queries x 64 dh).
    const unsigned short* qrow = Q + (size_t)(q0 + l16) * HDIM;
    v16bf qa0 = load_frag(qrow + ka0,      qrow + ka1);        // dh 0..31
    v16bf qa1 = load_frag(qrow + 32 + ka0, qrow + 32 + ka1);   // dh 32..63

    float m[8], l[8];
    v8f o[4] = {{}, {}, {}, {}};
#pragma unroll
    for (int r = 0; r < 8; ++r) { m[r] = -__builtin_inff(); l[r] = 0.f; }

    for (int kc = 0; kc < q0 + 16; kc += 32) {       // causal bound
        // ---- S = Q K^T for 32 keys (two 16-wide N tiles)
        const unsigned short* kr0 = K + (size_t)(kc + l16) * HDIM + kb2;
        const unsigned short* kr1 = K + (size_t)(kc + 16 + l16) * HDIM + kb2;
        v8f s0 = {}, s1 = {};
        s0 = wmma_bf16(qa0, load_frag(kr0, kr0 + 8), s0);
        s1 = wmma_bf16(qa0, load_frag(kr1, kr1 + 8), s1);
        s0 = wmma_bf16(qa1, load_frag(kr0 + 32, kr0 + 40), s0);
        s1 = wmma_bf16(qa1, load_frag(kr1 + 32, kr1 + 40), s1);

        // ---- online softmax (rows of C tile live in one 16-lane half)
        float alpha[8];
#pragma unroll
        for (int r = 0; r < 8; ++r) {
            int row = q0 + r + half * 8;
            float v0 = s0[r] * SOFTMAX_SCALE;
            float v1 = s1[r] * SOFTMAX_SCALE;
            if (kc + l16 > row)      v0 = -__builtin_inff();
            if (kc + 16 + l16 > row) v1 = -__builtin_inff();
            float t = fmaxf(v0, v1);
#pragma unroll
            for (int msk = 8; msk >= 1; msk >>= 1)
                t = fmaxf(t, __shfl_xor(t, msk, 16));
            float mn = fmaxf(m[r], t);
            float p0 = __expf(v0 - mn);
            float p1 = __expf(v1 - mn);
            alpha[r] = __expf(m[r] - mn);
            float sum = p0 + p1;
#pragma unroll
            for (int msk = 8; msk >= 1; msk >>= 1)
                sum += __shfl_xor(sum, msk, 16);
            l[r] = l[r] * alpha[r] + sum;
            m[r] = mn;
            P[(r + half * 8) * 32 + l16]      = f32_to_bf16(p0);
            P[(r + half * 8) * 32 + 16 + l16] = f32_to_bf16(p1);
        }
#pragma unroll
        for (int n = 0; n < 4; ++n)
#pragma unroll
            for (int r = 0; r < 8; ++r) o[n][r] *= alpha[r];

        __builtin_amdgcn_wave_barrier();             // LDS write -> read fence
        // ---- O += P V   (A-frag from LDS, B-frag contiguous from Vt)
        const unsigned short* prow = P + l16 * 32;
        v16bf pa = load_frag(prow + ka0, prow + ka1);
#pragma unroll
        for (int n = 0; n < 4; ++n) {
            const unsigned short* vc = Vt + (size_t)(n * 16 + l16) * SEQ + kc + kb2;
            o[n] = wmma_bf16(pa, load_frag(vc, vc + 8), o[n]);
        }
        __builtin_amdgcn_wave_barrier();             // LDS read done before next write
    }

    // ---- epilogue: normalize and store to [B*T, EMB] bf16 for projection
#pragma unroll
    for (int r = 0; r < 8; ++r) {
        float inv = 1.f / l[r];
        size_t row = (size_t)b * SEQ + q0 + r + half * 8;
#pragma unroll
        for (int n = 0; n < 4; ++n)
            attnb[row * EMB + h * HDIM + n * 16 + l16] = f32_to_bf16(o[n][r] * inv);
    }
}

// ---------------------------------------------------------------- projection
// attnb[ROWS][EMB] (bf16) @ WprojT[EMB][EMB] (bf16) -> out fp32.
// 16x64 output tile per wave, shared A-fragment.
__global__ void __launch_bounds__(256)
k_proj_gemm(const unsigned short* __restrict__ ab,
            const unsigned short* __restrict__ wT,
            float* __restrict__ out) {
    const int NCT = EMB / 64;                        // 6 col super-tiles
    int wave = (blockIdx.x * blockDim.x + threadIdx.x) >> 5;
    int lane = threadIdx.x & 31;
    int tr = wave / NCT, tc = wave - tr * NCT;
    if (tr >= ROWS / 16) return;
    int half = lane >> 4, l16 = lane & 15;
    int ka0 = half * 16, ka1 = half * 16 + 8, kb2 = half * 16;

    const unsigned short* arow = ab + (size_t)(tr * 16 + l16) * EMB;
    const unsigned short* bc0  = wT + (size_t)(tc * 64 + l16) * EMB + kb2;

    v8f acc[4] = {{}, {}, {}, {}};
#pragma unroll
    for (int k = 0; k < EMB; k += 32) {
        v16bf a = load_frag(arow + k + ka0, arow + k + ka1);
#pragma unroll
        for (int n = 0; n < 4; ++n) {
            const unsigned short* bp = bc0 + (size_t)(n * 16) * EMB + k;
            acc[n] = wmma_bf16(a, load_frag(bp, bp + 8), acc[n]);
        }
    }
#pragma unroll
    for (int n = 0; n < 4; ++n)
#pragma unroll
        for (int r = 0; r < 8; ++r)
            out[(size_t)(tr * 16 + r + half * 8) * EMB + tc * 64 + n * 16 + l16] =
                acc[n][r];
}

// ---------------------------------------------------------------- launch
extern "C" void kernel_launch(void* const* d_in, const int* in_sizes, int n_in,
                              void* d_out, int out_size, void* d_ws, size_t ws_size,
                              hipStream_t stream) {
    const float* x      = (const float*)d_in[0];   // [128,256,384]
    const float* W_qkv  = (const float*)d_in[1];   // [384,1152]
    const float* W_proj = (const float*)d_in[2];   // [384,384]
    float* out = (float*)d_out;

    // Workspace layout (bf16 stored as ushort). Total ~121 MB.
    unsigned short* ws     = (unsigned short*)d_ws;
    unsigned short* xb     = ws;                             // 32768*384
    unsigned short* wqkvT  = xb     + (size_t)ROWS * EMB;    // 1152*384
    unsigned short* wprojT = wqkvT  + (size_t)EMB3 * EMB;    // 384*384
    unsigned short* Qb     = wprojT + (size_t)EMB  * EMB;    // B*H*T*Dh
    unsigned short* Kb     = Qb     + (size_t)ROWS * EMB;
    unsigned short* Vtb    = Kb     + (size_t)ROWS * EMB;
    unsigned short* attnb  = Vtb    + (size_t)ROWS * EMB;

    // 1) conversions
    {
        int n = ROWS * EMB;
        k_f32_to_bf16<<<(n + 255) / 256, 256, 0, stream>>>(x, xb, n);
        int nq = EMB * EMB3;
        k_transpose_bf16<<<(nq + 255) / 256, 256, 0, stream>>>(W_qkv, wqkvT, EMB, EMB3);
        int np = EMB * EMB;
        k_transpose_bf16<<<(np + 255) / 256, 256, 0, stream>>>(W_proj, wprojT, EMB, EMB);
    }
    // 2) QKV projection (2048 x 18 super-tiles, 1 wave each, 8 waves/block)
    {
        int waves = (ROWS / 16) * (EMB3 / 64);     // 36864
        k_qkv_gemm<<<waves / 8, 256, 0, stream>>>(xb, wqkvT, Qb, Kb, Vtb);
    }
    // 3) fused causal flash attention (128*6*16 waves)
    {
        int waves = BATCH * NHEAD * (SEQ / 16);    // 12288
        k_attention<<<waves / 8, 256, 0, stream>>>(Qb, Kb, Vtb, attnb);
    }
    // 4) output projection (2048 x 6 super-tiles)
    {
        int waves = (ROWS / 16) * (EMB / 64);      // 12288
        k_proj_gemm<<<waves / 8, 256, 0, stream>>>(attnb, wprojT, out);
    }
}
